// PatchBasedInpainting_10703058501861
// MI455X (gfx1250) — matrix operand order repaired and checked by reading, not compiled, not run on hardware
//
#include <hip/hip_runtime.h>
#include <hip/hip_bf16.h>

#define H 2160
#define W 3840
#define HW (H * W)
#define NPATCH 256
#define P 128
#define NBLK 1024
#define EPS 1e-8f
#define LSTR 144   // LDS row stride in halfs: 9*16 -> every row base 16B aligned

typedef __attribute__((ext_vector_type(16))) _Float16 v16h;
typedef __attribute__((ext_vector_type(8)))  _Float16 v8h;
typedef __attribute__((ext_vector_type(4)))  _Float16 v4h;
typedef __attribute__((ext_vector_type(2)))  _Float16 v2h;
typedef __attribute__((ext_vector_type(8)))  float    v8f;

// -------------------------------------------------------------------------
// Kernel 1: feather masks via WMMA. One block per patch (256 threads = 8
// waves). Separable 9-tap Gaussian as two banded 16x16x32 f16 WMMA passes.
// All LDS traffic is unconditional vector ops; halo zero-fill is done with
// wave-uniform branches so no per-lane predication is generated.
// -------------------------------------------------------------------------
__global__ __launch_bounds__(256) void feather_kernel(
    const float* __restrict__ mask,
    const int* __restrict__ pos_y, const int* __restrict__ pos_x,
    float* __restrict__ fm)
{
    // phase A: holes[row 0..127][col+4, stride 144]
    // phase B: tmpT [col 0..127][row+4, stride 144]   (transposed tmp)
    __shared__ __align__(16) _Float16 smem[128 * LSTR];
    __shared__ float gsh[9];

    const int tid = threadIdx.x;
    const int n = blockIdx.x;

    if (tid == 0) {
        float g[9]; float s = 0.f;
        #pragma unroll
        for (int t = 0; t < 9; ++t) {
            float d = (float)(t - 4) * (3.0f / 4.0f);   // sigma = 4/3
            g[t] = __expf(-0.5f * d * d); s += g[t];
        }
        #pragma unroll
        for (int t = 0; t < 9; ++t) gsh[t] = g[t] / s;
    }

    const v4h z4 = {};
    const v8h z8 = {};

    // halo zero-fill, wave-uniform split: waves 0-3 low pad, waves 4-7 high pad
    if (tid < 128) {
        *(v4h*)(smem + tid * LSTR) = z4;                 // cols 0-3
    } else {
        _Float16* rp = smem + (tid - 128) * LSTR;
        *(v4h*)(rp + 132) = z4;                          // cols 132-135
        *(v8h*)(rp + 136) = z8;                          // cols 136-143
    }
    const int py0 = pos_y[n], px0 = pos_x[n];
    // interior gather: unconditional, 2 halfs packed per ds_store_b32
    for (int idx = tid; idx < 128 * 64; idx += 256) {
        int r = idx >> 6, c2 = (idx & 63) * 2;
        const float* mp = &mask[(size_t)(py0 + r) * W + px0 + c2];
        v2h pk;
        pk[0] = (_Float16)mp[0];
        pk[1] = (_Float16)mp[1];
        *(v2h*)(smem + r * LSTR + 4 + c2) = pk;
    }
    __syncthreads();

    const int lane = tid & 31;
    const int wv   = tid >> 5;     // wave id 0..7 (8 tiles per pass each)
    const int hp   = lane >> 4;    // lane half
    const int lrow = lane & 15;

    // constant B fragment, horizontal pass: B[K][c] = g[K - c], 0 <= K-c <= 8
    v16h bh;
    #pragma unroll
    for (int e = 0; e < 16; ++e) {
        int K = e + ((e < 8) ? 0 : 8) + hp * 8;
        int d = K - lrow;
        bh[e] = (d >= 0 && d <= 8) ? (_Float16)gsh[d] : (_Float16)0.f;
    }
    // constant A fragment, vertical pass: A[r][K] = g[K - r]
    v16h av;
    #pragma unroll
    for (int e = 0; e < 16; ++e) {
        int K = e + hp * 16;
        int d = K - lrow;
        av[e] = (d >= 0 && d <= 8 && K < 24) ? (_Float16)gsh[d] : (_Float16)0.f;
    }

    // ---- horizontal pass: acc[i] = In(16x32) x Band(32x16), kept in regs ----
    v8f acc[8];
    #pragma unroll
    for (int i = 0; i < 8; ++i) {
        int t = wv * 8 + i, ty = t >> 3, tx = t & 7;
        const _Float16* base = smem + (ty * 16 + lrow) * LSTR + tx * 16 + hp * 8;
        v8h lo = *(const v8h*)(base);         // K = hp*8   .. hp*8+7
        v8h hi = *(const v8h*)(base + 16);    // K = 16+hp*8.. 16+hp*8+7
        v16h a = __builtin_shufflevector(lo, hi,
            0, 1, 2, 3, 4, 5, 6, 7, 8, 9, 10, 11, 12, 13, 14, 15);
        v8f cz = {};
        acc[i] = __builtin_amdgcn_wmma_f32_16x16x32_f16(
            false, a, false, bh, (short)0, cz, false, false);
    }
    __syncthreads();

    // repurpose smem as tmpT[col][row+4]: halo zero-fill, wave-uniform split
    if (tid < 128) {
        *(v4h*)(smem + tid * LSTR) = z4;                 // rows -4..-1
    } else {
        _Float16* rp = smem + (tid - 128) * LSTR;
        *(v4h*)(rp + 132) = z4;                          // rows 128..131
        *(v8h*)(rp + 136) = z8;                          // rows 132..139 (pad)
    }
    #pragma unroll
    for (int i = 0; i < 8; ++i) {
        int t = wv * 8 + i, ty = t >> 3, tx = t & 7;
        v4h lo4, hi4;
        #pragma unroll
        for (int v = 0; v < 4; ++v) {
            lo4[v] = (_Float16)acc[i][v];
            hi4[v] = (_Float16)acc[i][v + 4];
        }
        // tmp value (row = ty*16+v+hp*8, col = tx*16+lrow) -> tmpT[col][row+4]
        _Float16* dst = smem + (tx * 16 + lrow) * LSTR + ty * 16 + hp * 8 + 4;
        *(v4h*)(dst)     = lo4;   // 8B aligned
        *(v4h*)(dst + 4) = hi4;   // 16B aligned
    }
    __syncthreads();

    // ---- vertical pass: Out = Band^T(16x32) x Tmp(32x16) ----
    #pragma unroll
    for (int i = 0; i < 8; ++i) {
        int t = wv * 8 + i, ty = t >> 3, tx = t & 7;
        // B[K][c] = tmp[ty*16-4+K][tx*16+c] = tmpT[tx*16+c][ty*16+K]
        const _Float16* base = smem + (tx * 16 + lrow) * LSTR + ty * 16 + hp * 16;
        v8h lo = *(const v8h*)(base);         // K = hp*16   .. hp*16+7
        v8h hi = *(const v8h*)(base + 8);     // K = hp*16+8 .. hp*16+15
        v16h b = __builtin_shufflevector(lo, hi,
            0, 1, 2, 3, 4, 5, 6, 7, 8, 9, 10, 11, 12, 13, 14, 15);
        v8f cz = {};
        v8f o = __builtin_amdgcn_wmma_f32_16x16x32_f16(
            false, av, false, b, (short)0, cz, false, false);
        #pragma unroll
        for (int v = 0; v < 8; ++v)
            fm[((size_t)n * P + ty * 16 + v + hp * 8) * P + tx * 16 + lrow] = o[v];
    }
}

// -------------------------------------------------------------------------
// Kernel 2: ordered per-pixel blend (equivalent to the sequential scan).
// 32x8 tile per block; block-uniform bbox reject against each patch.
// -------------------------------------------------------------------------
__global__ __launch_bounds__(256) void scatter_kernel(
    const float* __restrict__ orig, const float* __restrict__ patches,
    const float* __restrict__ fm,
    const int* __restrict__ pos_y, const int* __restrict__ pos_x,
    float* __restrict__ out)
{
    __shared__ int spy[NPATCH], spx[NPATCH];
    const int tid = threadIdx.x;
    spy[tid] = pos_y[tid];
    spx[tid] = pos_x[tid];
    __syncthreads();

    const int x0 = blockIdx.x * 32, y0 = blockIdx.y * 8;
    const int px = x0 + (tid & 31), py = y0 + (tid >> 5);
    const size_t pix = (size_t)py * W + px;

    float v0 = orig[pix];
    float v1 = orig[(size_t)HW + pix];
    float v2 = orig[2 * (size_t)HW + pix];

    for (int i = 0; i < NPATCH; ++i) {
        int yy = spy[i], xx = spx[i];
        // block-uniform overlap test against this 32x8 tile
        if (yy < y0 + 8 && yy + P > y0 && xx < x0 + 32 && xx + P > x0) {
            int dy = py - yy, dx = px - xx;
            if ((unsigned)dy < (unsigned)P && (unsigned)dx < (unsigned)P) {
                float f = fm[((size_t)i * P + dy) * P + dx];
                float omf = 1.f - f;
                const float* p = patches + (size_t)i * 3 * P * P + dy * P + dx;
                v0 = v0 * omf + p[0] * f;
                v1 = v1 * omf + p[P * P] * f;
                v2 = v2 * omf + p[2 * P * P] * f;
            }
        }
    }
    out[pix] = v0;
    out[(size_t)HW + pix] = v1;
    out[2 * (size_t)HW + pix] = v2;
}

// -------------------------------------------------------------------------
// Kernel 3: single-pass masked moments (13 sums), deterministic two-stage.
// -------------------------------------------------------------------------
__global__ __launch_bounds__(256) void stats_kernel(
    const float* __restrict__ orig, const float* __restrict__ fused,
    const float* __restrict__ mask, float* __restrict__ partials)
{
    const size_t stride = (size_t)NBLK * 256;
    float s[13];
    #pragma unroll
    for (int v = 0; v < 13; ++v) s[v] = 0.f;

    for (size_t idx = (size_t)blockIdx.x * 256 + threadIdx.x; idx < (size_t)HW;
         idx += stride) {
        // pull next stride's lines toward L2 (global_prefetch_b8)
        if (idx + stride < (size_t)HW) {
            __builtin_prefetch(&mask[idx + stride], 0, 1);
            __builtin_prefetch(&orig[idx + stride], 0, 1);
            __builtin_prefetch(&fused[idx + stride], 0, 1);
        }
        float nh = 1.f - mask[idx];
        s[0] += nh;
        #pragma unroll
        for (int c = 0; c < 3; ++c) {
            float o = orig[(size_t)c * HW + idx];
            float f = fused[(size_t)c * HW + idx];
            s[1 + c]  += o * nh;
            s[4 + c]  += o * o * nh;
            s[7 + c]  += f * nh;
            s[10 + c] += f * f * nh;
        }
    }

    __shared__ float red[256];
    #pragma unroll
    for (int v = 0; v < 13; ++v) {
        red[threadIdx.x] = s[v];
        __syncthreads();
        for (int off = 128; off > 0; off >>= 1) {
            if (threadIdx.x < off) red[threadIdx.x] += red[threadIdx.x + off];
            __syncthreads();
        }
        if (threadIdx.x == 0) partials[(size_t)blockIdx.x * 16 + v] = red[0];
        __syncthreads();
    }
}

__global__ void finalize_kernel(const float* __restrict__ partials,
                                float* __restrict__ params)
{
    __shared__ float tot[13];
    int v = threadIdx.x;
    if (v < 13) {
        float s = 0.f;
        for (int b = 0; b < NBLK; ++b) s += partials[(size_t)b * 16 + v];
        tot[v] = s;
    }
    __syncthreads();
    if (v == 0) {
        float denom = tot[0];
        #pragma unroll
        for (int c = 0; c < 3; ++c) {
            float om   = tot[1 + c] / denom;
            float os   = sqrtf(fmaxf(tot[4 + c] / denom - om * om, 0.f));
            float fmn  = tot[7 + c] / denom;
            float fs   = sqrtf(fmaxf(tot[10 + c] / denom - fmn * fmn, 0.f));
            params[c]     = fmn;             // fused hole mean
            params[3 + c] = os / (fs + EPS); // scale
            params[6 + c] = om;              // original mean
        }
    }
}

// -------------------------------------------------------------------------
// Kernel 4: in-place color correction of d_out.
// -------------------------------------------------------------------------
__global__ __launch_bounds__(256) void correct_kernel(
    const float* __restrict__ mask, const float* __restrict__ params,
    float* __restrict__ img)
{
    size_t idx = (size_t)blockIdx.x * 256 + threadIdx.x;
    if (idx >= (size_t)3 * HW) return;
    int c = (int)(idx / HW);
    size_t pix = idx % HW;
    float m = mask[pix];
    float f = img[idx];
    float hole = f * m;
    float corr = (hole - params[c]) * params[3 + c] + params[6 + c];
    img[idx] = f * (1.f - m) + corr * m;
}

extern "C" void kernel_launch(void* const* d_in, const int* in_sizes, int n_in,
                              void* d_out, int out_size, void* d_ws, size_t ws_size,
                              hipStream_t stream) {
    const float* orig    = (const float*)d_in[0];
    const float* patches = (const float*)d_in[1];
    const float* mask    = (const float*)d_in[2];
    const int*   pos_y   = (const int*)d_in[3];
    const int*   pos_x   = (const int*)d_in[4];
    float* out = (float*)d_out;

    float* fm       = (float*)d_ws;                       // 256*128*128 floats
    float* partials = fm + (size_t)NPATCH * P * P;        // NBLK*16 floats
    float* params   = partials + (size_t)NBLK * 16;       // 16 floats

    feather_kernel<<<NPATCH, 256, 0, stream>>>(mask, pos_y, pos_x, fm);
    scatter_kernel<<<dim3(W / 32, H / 8), 256, 0, stream>>>(orig, patches, fm,
                                                            pos_y, pos_x, out);
    stats_kernel<<<NBLK, 256, 0, stream>>>(orig, out, mask, partials);
    finalize_kernel<<<1, 64, 0, stream>>>(partials, params);
    correct_kernel<<<(int)((3 * (size_t)HW + 255) / 256), 256, 0, stream>>>(
        mask, params, out);
}
